// Kmeans_80977313399780
// MI455X (gfx1250) — matrix operand order repaired and checked by reading, not compiled, not run on hardware
//
#include <hip/hip_runtime.h>
#include <hip/hip_bf16.h>

typedef __attribute__((ext_vector_type(16))) _Float16 v16h;
typedef __attribute__((ext_vector_type(8)))  _Float16 v8h;
typedef __attribute__((ext_vector_type(8)))  float    v8f;

#define BN 65536
#define KN 1024
#define DN 256

// ---------------------------------------------------------------------------
// Kernel 1: pack centroids f32 [K,D] -> f16 B-fragment layout, compute c_sq.
// Layout: bp[((d/32)*KN + k)*32 + (d%32)] = f16(C[k][d])
// so a lane (col n, half h) reads 8 halves at ((kc*KN+n)*32 + h*8) and 8 at +16.
// ---------------------------------------------------------------------------
__global__ void pack_centroids_kernel(const float* __restrict__ cents,
                                      float* __restrict__ csq,
                                      _Float16* __restrict__ bp) {
    const int n = blockIdx.x;      // centroid index
    const int t = threadIdx.x;     // d index, blockDim == DN == 256
    const float v = cents[(size_t)n * DN + t];
    bp[((size_t)(t >> 5) * KN + n) * 32 + (t & 31)] = (_Float16)v;

    __shared__ float red[DN];
    red[t] = v * v;
    __syncthreads();
    #pragma unroll
    for (int s = DN / 2; s > 0; s >>= 1) {
        if (t < s) red[t] += red[t + s];
        __syncthreads();
    }
    if (t == 0) csq[n] = red[0];
}

// ---------------------------------------------------------------------------
// Kernel 2: per-wave 16-row tile; WMMA f32<=f16 GEMM vs all 1024 centroids,
// running argmin, cross-lane reduce, index write + centroid gather.
// ---------------------------------------------------------------------------
__global__ void __launch_bounds__(256)
assign_kernel(const float* __restrict__ x,
              const float* __restrict__ cents,
              const float* __restrict__ csq,
              const _Float16* __restrict__ bp,
              float* __restrict__ out) {
    const int lane    = threadIdx.x & 31;
    const int wave    = threadIdx.x >> 5;
    const int rowBase = (blockIdx.x * 8 + wave) * 16;
    const int m       = lane & 15;   // row (A) / column (B,C) within tile
    const int h       = lane >> 4;   // lane-half selects K sub-range

    // ---- Load A fragments once: 16 rows x 256 d, f32 -> f16 --------------
    // 16-bit A 16x32 layout: lanes 0-15 hold K {0..7} in v0-3, {16..23} in
    // v4-7; lanes 16-31 hold {8..15} and {24..31}.  h*8 gives the split.
    v16h a[8];
    const float* xr = x + (size_t)(rowBase + m) * DN;
    #pragma unroll
    for (int kc = 0; kc < 8; ++kc) {
        const float4 l0 = *(const float4*)(xr + kc * 32 + h * 8);
        const float4 l1 = *(const float4*)(xr + kc * 32 + h * 8 + 4);
        const float4 h0 = *(const float4*)(xr + kc * 32 + 16 + h * 8);
        const float4 h1 = *(const float4*)(xr + kc * 32 + 16 + h * 8 + 4);
        a[kc][0]  = (_Float16)l0.x; a[kc][1]  = (_Float16)l0.y;
        a[kc][2]  = (_Float16)l0.z; a[kc][3]  = (_Float16)l0.w;
        a[kc][4]  = (_Float16)l1.x; a[kc][5]  = (_Float16)l1.y;
        a[kc][6]  = (_Float16)l1.z; a[kc][7]  = (_Float16)l1.w;
        a[kc][8]  = (_Float16)h0.x; a[kc][9]  = (_Float16)h0.y;
        a[kc][10] = (_Float16)h0.z; a[kc][11] = (_Float16)h0.w;
        a[kc][12] = (_Float16)h1.x; a[kc][13] = (_Float16)h1.y;
        a[kc][14] = (_Float16)h1.z; a[kc][15] = (_Float16)h1.w;
    }

    float minv[8];
    int   mini[8];
    #pragma unroll
    for (int i = 0; i < 8; ++i) { minv[i] = 3.4e38f; mini[i] = 0; }

    // ---- Sweep all 64 N-tiles of 16 centroids ----------------------------
    for (int nt = 0; nt < KN / 16; ++nt) {
        const int n = nt * 16 + m;           // this lane's centroid column
        v8f acc = {};

        v16h bfrag[8];
        #pragma unroll
        for (int kc = 0; kc < 8; ++kc) {
            const v8h* p = (const v8h*)(bp + ((size_t)kc * KN + n) * 32 + h * 8);
            const v8h lo = p[0];
            const v8h hi = p[2];
            #pragma unroll
            for (int j = 0; j < 8; ++j) { bfrag[kc][j] = lo[j]; bfrag[kc][8 + j] = hi[j]; }
        }
        #pragma unroll
        for (int kc = 0; kc < 8; ++kc) {
            acc = __builtin_amdgcn_wmma_f32_16x16x32_f16(
                false, a[kc], false, bfrag[kc], (short)0, acc, false, false);
        }

        const float cs = csq[n];
        #pragma unroll
        for (int i = 0; i < 8; ++i) {
            const float d = cs - 2.0f * acc[i];   // ||c||^2 - 2 x.c
            if (d < minv[i]) { minv[i] = d; mini[i] = n; }
        }
    }

    // ---- Cross-lane argmin within each 16-lane group ---------------------
    // C/D layout: lanes 0-15 hold rows rowBase+0..7 (VGPR i -> row i),
    // lanes 16-31 hold rows rowBase+8..15.
    #pragma unroll
    for (int i = 0; i < 8; ++i) {
        #pragma unroll
        for (int off = 1; off < 16; off <<= 1) {
            const float ov = __shfl_xor(minv[i], off, 16);
            const int   oi = __shfl_xor(mini[i], off, 16);
            if (ov < minv[i] || (ov == minv[i] && oi < mini[i])) {
                minv[i] = ov; mini[i] = oi;
            }
        }
    }

    // ---- Write indices (as float, matching d_out dtype) ------------------
    float* out_idx = out;
    if (lane == 0) {
        #pragma unroll
        for (int i = 0; i < 8; ++i) out_idx[rowBase + i] = (float)mini[i];
    } else if (lane == 16) {
        #pragma unroll
        for (int i = 0; i < 8; ++i) out_idx[rowBase + 8 + i] = (float)mini[i];
    }

    // ---- Gather centroids[idx] : 16 rows x 1 KB, whole wave per row ------
    float* out_g = out + BN;
    #pragma unroll
    for (int r = 0; r < 16; ++r) {
        const int src = (r < 8) ? 0 : 16;
        const int bi  = __shfl(mini[r & 7], src, 32);
        const float* crow = cents + (size_t)bi * DN;
        float*       orow = out_g + (size_t)(rowBase + r) * DN;
        *(float4*)(orow + lane * 8)     = *(const float4*)(crow + lane * 8);
        *(float4*)(orow + lane * 8 + 4) = *(const float4*)(crow + lane * 8 + 4);
    }
}

// ---------------------------------------------------------------------------
extern "C" void kernel_launch(void* const* d_in, const int* in_sizes, int n_in,
                              void* d_out, int out_size, void* d_ws, size_t ws_size,
                              hipStream_t stream) {
    const float* x     = (const float*)d_in[0];   // [B, D] f32
    const float* cents = (const float*)d_in[1];   // [K, D] f32
    float* out = (float*)d_out;                   // [B] idx-as-float ++ [B, D] gathered

    // Workspace: c_sq (K floats, 4 KB) then packed f16 centroids (512 KB).
    float*    csq = (float*)d_ws;
    _Float16* bp  = (_Float16*)((char*)d_ws + 4096);

    pack_centroids_kernel<<<KN, DN, 0, stream>>>(cents, csq, bp);

    // 256 threads = 8 waves; each wave owns 16 rows -> 128 rows per block.
    assign_kernel<<<BN / 128, 256, 0, stream>>>(x, cents, csq, bp, out);
}
